// OFFLINE_AE_RPN_20074677141677
// MI455X (gfx1250) — compile-verified, matrix operand order checked by reading, not run on hardware
//
#include <hip/hip_runtime.h>
#include <hip/hip_bf16.h>

typedef __attribute__((ext_vector_type(16))) _Float16 v16h;
typedef __attribute__((ext_vector_type(8)))  float    v8f;

namespace {

constexpr int kA  = 261888;   // total anchors (5 FPN levels x 3 ratios)
constexpr int kB  = 4;
constexpr int kG  = 64;
constexpr int kBA = kB * kA;
static_assert(kA % 256 == 0, "anchor count must tile by 256");

// Deterministic per-(batch, anchor) 32-bit random key (threefry stand-in).
__device__ __forceinline__ unsigned hashrand(unsigned b, unsigned a) {
  unsigned x = a * 0x9E3779B9u + (b + 1u) * 0x85EBCA6Bu;
  x ^= x >> 16; x *= 0x7FEB352Du;
  x ^= x >> 15; x *= 0x846CA68Bu;
  x ^= x >> 16;
  return x;
}

__device__ __forceinline__ float iou_box(const float4 g, float ga,
                                         const float4 a, float aa) {
  float lx = fmaxf(g.x, a.x);
  float ly = fmaxf(g.y, a.y);
  float rx = fminf(g.z, a.z);
  float ry = fminf(g.w, a.w);
  float w  = fmaxf(rx - lx, 0.0f);
  float h  = fmaxf(ry - ly, 0.0f);
  float inter = w * h;
  return inter / fmaxf(ga + aa - inter, 1e-6f);
}

} // namespace

// ---------------------------------------------------------------------------
// Kernel 0: zero workspace state + loss accumulator (replay-safe).
// ---------------------------------------------------------------------------
__global__ void k_init(int* __restrict__ hi, int* __restrict__ hist,
                       int* __restrict__ state, float* __restrict__ out_loss) {
  int t = blockIdx.x * blockDim.x + threadIdx.x;
  if (t < kB * kG) hi[t] = 0;
  for (int i = t; i < kB * 512; i += gridDim.x * blockDim.x) hist[i] = 0;
  if (t < kB * 2 * 3) state[t] = 0;
  if (t == 0) *out_loss = 0.0f;
}

// ---------------------------------------------------------------------------
// Kernel 1: fused IoU + matcher. One thread per anchor; GT boxes staged in
// LDS. Per-gt maxima use a read-filtered LDS atomicMax: broadcast ds_load of
// smax[g] (all lanes same address), atomic only when this lane beats it —
// IoU sparsity makes the atomic rare. Float bits compared as ints (IoU >= 0).
// ---------------------------------------------------------------------------
__global__ void k_iou_match(const float* __restrict__ anchors,
                            const float* __restrict__ gt_boxes,
                            const int*   __restrict__ gt_classes,
                            float* __restrict__ out_midx,
                            float* __restrict__ out_boxes,
                            float* __restrict__ out_mcls,
                            int*   __restrict__ lab,
                            int*   __restrict__ hi) {
  __shared__ float4 sgt[kG];
  __shared__ float  sarea[kG];
  __shared__ float  scls[kG];
  __shared__ int    smax[kG];
  const int b = blockIdx.y;
  const int t = threadIdx.x;
  if (t < kG) {
    float4 g = ((const float4*)gt_boxes)[b * kG + t];
    sgt[t]   = g;
    sarea[t] = (g.z - g.x) * (g.w - g.y);
    scls[t]  = (float)gt_classes[b * kG + t];
    smax[t]  = 0;
  }
  __syncthreads();

  const int a = blockIdx.x * blockDim.x + t;
  float4 av  = ((const float4*)anchors)[a];
  float aarea = (av.z - av.x) * (av.w - av.y);

  float best = -1.0f;
  int   bidx = 0;
  #pragma unroll 4
  for (int g = 0; g < kG; ++g) {
    float v = iou_box(sgt[g], sarea[g], av, aarea);
    if (v > best) { best = v; bidx = g; }
    int vb = __float_as_int(v);
    if (vb > smax[g]) atomicMax(&smax[g], vb);   // rare: filtered by read
  }
  __syncthreads();
  if (t < kG && smax[t] != 0) atomicMax(&hi[b * kG + t], smax[t]);

  const int idx = b * kA + a;
  lab[idx]      = (best < 0.3f) ? 0 : ((best >= 0.7f) ? 1 : -1);
  out_midx[idx] = (float)bidx;
  ((float4*)out_boxes)[idx] = sgt[bidx];
  out_mcls[idx] = scls[bidx];
}

// ---------------------------------------------------------------------------
// Kernel 2: low-quality-match pass (IoU recomputed; cheaper than 268 MB of
// IoU traffic), finalize labels, build level-24 radix histograms.
// ---------------------------------------------------------------------------
__global__ void k_lq_hist(const float* __restrict__ anchors,
                          const float* __restrict__ gt_boxes,
                          const int*   __restrict__ hi,
                          int*   __restrict__ lab,
                          int*   __restrict__ hist) {
  __shared__ float4 sgt[kG];
  __shared__ float  sarea[kG];
  __shared__ float  shi[kG];
  const int b = blockIdx.y;
  const int t = threadIdx.x;
  if (t < kG) {
    float4 g = ((const float4*)gt_boxes)[b * kG + t];
    sgt[t]   = g;
    sarea[t] = (g.z - g.x) * (g.w - g.y);
    shi[t]   = __int_as_float(hi[b * kG + t]);
  }
  __syncthreads();

  const int a = blockIdx.x * blockDim.x + t;
  float4 av  = ((const float4*)anchors)[a];
  float aarea = (av.z - av.x) * (av.w - av.y);

  bool lq = false;
  #pragma unroll 4
  for (int g = 0; g < kG; ++g) {
    float v  = iou_box(sgt[g], sarea[g], av, aarea);
    float hg = shi[g];
    lq = lq || ((hg > 0.0f) && (v >= hg));
  }
  const int idx = b * kA + a;
  int l = lab[idx];
  if (lq) l = 1;
  lab[idx] = l;
  if (l >= 0) {
    int cls = (l == 1) ? 0 : 1;
    unsigned h = hashrand(b, a);
    atomicAdd(&hist[(b * 2 + cls) * 256 + (h >> 24)], 1);
  }
}

// ---------------------------------------------------------------------------
// Kernel 3: radix-select scan step (tiny). At shift==24 also derives
// n_pos = min(128,|pos|), n_neg = min(256-n_pos,|neg|). Zeroes histograms
// for the next level.  state[b][cls] = {prefix, k_rem, tie_quota}.
// ---------------------------------------------------------------------------
__global__ void k_scan(int* __restrict__ hist, int* __restrict__ state,
                       int shift) {
  const int b = blockIdx.x;
  if (threadIdx.x == 0) {
    int nsel[2] = {0, 0};
    if (shift == 24) {
      int tp = 0, tn = 0;
      for (int i = 0; i < 256; ++i) {
        tp += hist[(b * 2 + 0) * 256 + i];
        tn += hist[(b * 2 + 1) * 256 + i];
      }
      nsel[0] = min(128, tp);
      nsel[1] = min(256 - nsel[0], tn);
    }
    for (int cls = 0; cls < 2; ++cls) {
      int* st = state + (b * 2 + cls) * 3;
      unsigned prefix = (shift == 24) ? 0u : (unsigned)st[0];
      int k = (shift == 24) ? nsel[cls] : st[1];
      const int* h = hist + (b * 2 + cls) * 256;
      int cum = 0, bucket = 0;
      for (int t = 255; t >= 0; --t) {
        int c = h[t];
        if (cum + c >= k) { bucket = t; break; }
        cum += c;
      }
      k -= cum;
      prefix |= ((unsigned)bucket) << shift;
      st[0] = (int)prefix;
      st[1] = k;
      if (shift == 0) st[2] = k;  // tie quota at exact threshold
    }
  }
  __syncthreads();
  for (int i = threadIdx.x; i < 512; i += blockDim.x) hist[b * 512 + i] = 0;
}

// ---------------------------------------------------------------------------
// Kernel 4: radix refine (shift = 16, 8, 0): histogram next byte for items
// matching the current prefix. Only touches the 4.2 MB label array.
// ---------------------------------------------------------------------------
__global__ void k_radix(const int* __restrict__ lab, int* __restrict__ hist,
                        const int* __restrict__ state, int shift) {
  const int b = blockIdx.y;
  const int a = blockIdx.x * blockDim.x + threadIdx.x;
  const int l = lab[b * kA + a];
  if (l < 0) return;
  const int cls = (l == 1) ? 0 : 1;
  const unsigned h = hashrand(b, a);
  const unsigned prefix = (unsigned)state[(b * 2 + cls) * 3];
  const unsigned maskhi = 0xFFFFFFFFu << (shift + 8);
  if ((h & maskhi) != (prefix & maskhi)) return;
  atomicAdd(&hist[(b * 2 + cls) * 256 + ((h >> shift) & 255)], 1);
}

// ---------------------------------------------------------------------------
// Kernel 5: final subsampled label write-out. key > T selected; key == T
// consumes the tie quota (keys are 32-bit random: ties are ~unique).
// ---------------------------------------------------------------------------
__global__ void k_final(const int* __restrict__ lab, int* __restrict__ state,
                        float* __restrict__ out_labels) {
  const int b = blockIdx.y;
  const int a = blockIdx.x * blockDim.x + threadIdx.x;
  const int idx = b * kA + a;
  const int l = lab[idx];
  float outl = -1.0f;
  if (l >= 0) {
    const int cls = (l == 1) ? 0 : 1;
    int* st = state + (b * 2 + cls) * 3;
    const unsigned h = hashrand(b, a);
    const unsigned T = (unsigned)st[0];
    bool sel = (h > T);
    if (!sel && h == T) sel = (atomicSub(&st[2], 1) > 0);
    if (sel) outl = (l == 1) ? 1.0f : 0.0f;
  }
  out_labels[idx] = outl;
}

// ---------------------------------------------------------------------------
// Kernel 6: loss_ae = mean(err^2) on the matrix pipe.
// D = A(x^2, f16 16x32) x B(ones, 32x16) + C(f32): every D entry is a row
// partial-sum replicated over 16 columns -> total = sum(D)/16. 512 floats
// per WMMA per wave (349184 = 682*512, no tail); accumulation stays f32;
// wave-uniform loop bounds keep EXEC all-ones as WMMA requires.
// ---------------------------------------------------------------------------
__global__ void k_loss_wmma(const float* __restrict__ err,
                            float* __restrict__ out_loss, int n) {
  const int lane   = threadIdx.x & 31;
  const int wave   = (blockIdx.x * blockDim.x + threadIdx.x) >> 5;
  const int nwaves = (gridDim.x * blockDim.x) >> 5;

  v16h ones;
  #pragma unroll
  for (int i = 0; i < 16; ++i) ones[i] = (_Float16)1.0f;
  v8f acc = {};

  const int nchunks = n >> 9;  // n / 512
  for (int c = wave; c < nchunks; c += nwaves) {
    const float* p = err + (c << 9) + lane * 16;
    v16h av;
    #pragma unroll
    for (int i = 0; i < 16; ++i) {
      float x = p[i];
      av[i] = (_Float16)(x * x);
    }
    acc = __builtin_amdgcn_wmma_f32_16x16x32_f16(
        false, av, false, ones, (short)0, acc, false, false);
  }

  float s = acc[0] + acc[1] + acc[2] + acc[3] +
            acc[4] + acc[5] + acc[6] + acc[7];
  #pragma unroll
  for (int off = 16; off > 0; off >>= 1) s += __shfl_xor(s, off, 32);
  if (lane == 0) atomicAdd(out_loss, s * (1.0f / 16.0f) / (float)n);
}

// ---------------------------------------------------------------------------
extern "C" void kernel_launch(void* const* d_in, const int* in_sizes, int n_in,
                              void* d_out, int out_size, void* d_ws,
                              size_t ws_size, hipStream_t stream) {
  const float* anchors    = (const float*)d_in[0];  // [A,4]
  const float* gt_boxes   = (const float*)d_in[1];  // [B,G,4]
  const int*   gt_classes = (const int*)  d_in[2];  // [B,G]
  const float* recons     = (const float*)d_in[3];  // [B,P]
  const int n_err = in_sizes[3];

  float* out        = (float*)d_out;
  float* out_labels = out;             // [B,A]
  float* out_midx   = out + kBA;       // [B,A]
  float* out_boxes  = out + 2 * kBA;   // [B,A,4]
  float* out_mcls   = out + 6 * kBA;   // [B,A]
  float* out_loss   = out + 7 * kBA;   // [1]

  int* ws    = (int*)d_ws;
  int* hi    = ws;                 // [B*G]       per-gt max IoU (float bits)
  int* lab   = ws + kB * kG;       // [B*A]       labels
  int* hist  = lab + kBA;          // [B][2][256] radix histograms
  int* state = hist + kB * 512;    // [B][2]{prefix,k_rem,tie}

  const dim3 gridA(kA / 256, kB);

  k_init     <<<8, 256, 0, stream>>>(hi, hist, state, out_loss);
  k_iou_match<<<gridA, 256, 0, stream>>>(anchors, gt_boxes, gt_classes,
                                         out_midx, out_boxes, out_mcls,
                                         lab, hi);
  k_lq_hist  <<<gridA, 256, 0, stream>>>(anchors, gt_boxes, hi, lab, hist);
  k_scan     <<<kB, 256, 0, stream>>>(hist, state, 24);
  k_radix    <<<gridA, 256, 0, stream>>>(lab, hist, state, 16);
  k_scan     <<<kB, 256, 0, stream>>>(hist, state, 16);
  k_radix    <<<gridA, 256, 0, stream>>>(lab, hist, state, 8);
  k_scan     <<<kB, 256, 0, stream>>>(hist, state, 8);
  k_radix    <<<gridA, 256, 0, stream>>>(lab, hist, state, 0);
  k_scan     <<<kB, 256, 0, stream>>>(hist, state, 0);
  k_final    <<<gridA, 256, 0, stream>>>(lab, state, out_labels);
  k_loss_wmma<<<64, 256, 0, stream>>>(recons, out_loss, n_err);

  (void)n_in; (void)out_size; (void)ws_size;
}
